// CuteNetModel_86981677678830
// MI455X (gfx1250) — matrix operand order repaired and checked by reading, not compiled, not run on hardware
//
#include <hip/hip_runtime.h>
#include <hip/hip_bf16.h>
#include <math.h>

// ---------------------------------------------------------------------------
// Swin block on gfx1250: all GEMMs + attention via v_wmma_f32_16x16x32_f16.
// Fragment loads are vectorized (ds_load_b128) via fragment-major LDS layouts.
// ---------------------------------------------------------------------------

typedef __attribute__((ext_vector_type(8)))  _Float16 v8h;
typedef __attribute__((ext_vector_type(16))) _Float16 v16h;
typedef __attribute__((ext_vector_type(8)))  float    v8f;

#define Hc 56
#define Wc 56
#define Cc 192
#define NHc 6
#define WSc 7
#define SSc 3
#define Nwin 49            // tokens per window
#define NWIN_IMG 64        // windows per image (8x8)
#define Bc 32
#define TOK 3136           // H*W
#define MROWS 100352       // B * TOK
#define HIDDEN 768
#define QKVN 576
#define HEADD 32
#define SCALEF 0.17677669529663687f

#define WMMA16(a, b, c) \
  __builtin_amdgcn_wmma_f32_16x16x32_f16(false, (a), false, (b), (short)0, (c), false, false)

// A fragment (16x32 f16): row = lane&15, 16 contiguous halves at k-offset
// hi*16 (same k order as B). rowptr points at the 32-wide k-window.
__device__ __forceinline__ v16h frag_a(const _Float16* rowptr, int hi) {
  return *(const v16h*)(rowptr + hi * 16);
}

// ------------------- f32 -> f16 transpose (weights) -------------------------
// in:  a[K][N] row-major f32;  out: o[N][K] row-major f16 (B^T, fragment-major)
__global__ __launch_bounds__(256) void cvt_transpose_f16(const float* __restrict__ a,
                                                         _Float16* __restrict__ o,
                                                         int K, int N) {
  int i = blockIdx.x * 256 + threadIdx.x;
  if (i < K * N) {
    int k = i / N, n = i - k * N;
    o[(size_t)n * K + k] = (_Float16)a[i];
  }
}

// ------------------- LayerNorm (+ optional shift/window) --------------------
__global__ __launch_bounds__(256) void ln_kernel(const float* __restrict__ x,
                                                 const float* __restrict__ g,
                                                 const float* __restrict__ b,
                                                 _Float16* __restrict__ out,
                                                 int windowed) {
  int row  = blockIdx.x * 8 + (threadIdx.x >> 5);
  int lane = threadIdx.x & 31;
  if (row >= MROWS) return;

  int src;
  if (windowed) {
    int bidx = row / TOK;
    int rem  = row - bidx * TOK;
    int wi   = rem / Nwin;
    int n    = rem - wi * Nwin;
    int hs   = (wi >> 3) * WSc + n / WSc;   // shifted-grid coords
    int ws_  = (wi & 7)  * WSc + n % WSc;
    int h = hs + SSc; if (h >= Hc) h -= Hc; // source of roll(-SS)
    int w = ws_ + SSc; if (w >= Wc) w -= Wc;
    src = bidx * TOK + h * Wc + w;
  } else {
    src = row;
  }

  const float* xr = x + (size_t)src * Cc;
  float vals[6];
  float s = 0.f;
#pragma unroll
  for (int i = 0; i < 6; ++i) { vals[i] = xr[lane + i * 32]; s += vals[i]; }
#pragma unroll
  for (int o = 16; o > 0; o >>= 1) s += __shfl_xor(s, o, 32);
  float mu = s * (1.f / Cc);
  float vs = 0.f;
#pragma unroll
  for (int i = 0; i < 6; ++i) { float d = vals[i] - mu; vs += d * d; }
#pragma unroll
  for (int o = 16; o > 0; o >>= 1) vs += __shfl_xor(vs, o, 32);
  float inv = rsqrtf(vs * (1.f / Cc) + 1e-5f);
#pragma unroll
  for (int i = 0; i < 6; ++i) {
    int c = lane + i * 32;
    out[(size_t)row * Cc + c] = (_Float16)((vals[i] - mu) * inv * g[c] + b[c]);
  }
}

// ----------------------------- WMMA GEMM -----------------------------------
// C[M,N] = A[M,K] (f16 row-major) @ B[K,N] given as Bt[N][K] (f16 row-major).
// Block: 256 threads = 8 waves; tile 128(M) x 64(N), K-step 32, double-buffered.
// Wave (wm 0..3, wn 0..1) computes a 32x32 sub-tile = 4 WMMAs per K-step.
// mode 0: out_h = f16(acc + bias)            (gelu optional)
// mode 1: proj epilogue: window-reverse + un-shift + residual -> out_f (f32)
// mode 2: fc2 epilogue:  acc + bias + add_src -> out_f (f32)
__global__ __launch_bounds__(256) void gemm_wmma(
    const _Float16* __restrict__ A, const _Float16* __restrict__ Bt,
    const float* __restrict__ bias, int M, int N, int K,
    int mode, int gelu,
    _Float16* __restrict__ out_h, float* __restrict__ out_f,
    const float* __restrict__ add_src) {
  __shared__ __align__(32) _Float16 As[2][128 * 32];   // [m][k]
  __shared__ __align__(32) _Float16 Bs[2][64 * 32];    // [n][k]  fragment-major

  const int tid  = threadIdx.x;
  const int bm   = blockIdx.x * 128;
  const int bn   = blockIdx.y * 64;
  const int wave = tid >> 5;
  const int lane = tid & 31;
  const int wm   = wave & 3;
  const int wn   = wave >> 2;
  const int nl   = lane & 15;
  const int hi   = lane >> 4;

  v8f acc00 = {}, acc01 = {}, acc10 = {}, acc11 = {};
  const int KT = K >> 5;

  // staging: A 4096 halves (2 x v8h per thread), B 2048 halves (1 x v8h)
#define STAGE(k0, buf)                                                         \
  do {                                                                         \
    _Pragma("unroll")                                                          \
    for (int i = 0; i < 2; ++i) {                                              \
      int chunk = tid * 2 + i;                                                 \
      int r = chunk >> 2, c = (chunk & 3) * 8;                                 \
      *(v8h*)&As[buf][r * 32 + c] =                                            \
          *(const v8h*)(A + (size_t)(bm + r) * K + (k0) + c);                  \
    }                                                                          \
    {                                                                          \
      int r = tid >> 2, c = (tid & 3) * 8;                                     \
      *(v8h*)&Bs[buf][r * 32 + c] =                                            \
          *(const v8h*)(Bt + (size_t)(bn + r) * K + (k0) + c);                 \
    }                                                                          \
  } while (0)

  STAGE(0, 0);
  for (int kt = 0; kt < KT; ++kt) {
    __syncthreads();
    if (kt + 1 < KT) STAGE((kt + 1) * 32, (kt + 1) & 1);
    const _Float16* Ab = As[kt & 1];
    const _Float16* Bb = Bs[kt & 1];
    v16h a0 = frag_a(Ab + (wm * 32 + nl) * 32, hi);
    v16h a1 = frag_a(Ab + (wm * 32 + 16 + nl) * 32, hi);
    v16h b0 = *(const v16h*)(Bb + (wn * 32 + nl) * 32 + hi * 16);
    v16h b1 = *(const v16h*)(Bb + (wn * 32 + 16 + nl) * 32 + hi * 16);
    acc00 = WMMA16(a0, b0, acc00);
    acc01 = WMMA16(a0, b1, acc01);
    acc10 = WMMA16(a1, b0, acc10);
    acc11 = WMMA16(a1, b1, acc11);
  }
#undef STAGE

  const v8f* accs[2][2] = {{&acc00, &acc01}, {&acc10, &acc11}};
#pragma unroll
  for (int t = 0; t < 2; ++t) {
#pragma unroll
    for (int u = 0; u < 2; ++u) {
      const v8f av = *accs[t][u];
#pragma unroll
      for (int r = 0; r < 8; ++r) {
        int grow = bm + wm * 32 + t * 16 + r + hi * 8;
        int gcol = bn + wn * 32 + u * 16 + nl;
        float v = av[r] + bias[gcol];
        if (mode == 0) {
          if (gelu) v = 0.5f * v * (1.0f + erff(v * 0.70710678118654752f));
          out_h[(size_t)grow * N + gcol] = (_Float16)v;
        } else if (mode == 1) {
          // grow is in window order -> image order (+SS roll back) + residual
          int bidx = grow / TOK;
          int rem  = grow - bidx * TOK;
          int wi   = rem / Nwin;
          int n    = rem - wi * Nwin;
          int hs   = (wi >> 3) * WSc + n / WSc;
          int ws_  = (wi & 7)  * WSc + n % WSc;
          int h = hs + SSc; if (h >= Hc) h -= Hc;
          int w = ws_ + SSc; if (w >= Wc) w -= Wc;
          size_t idx = ((size_t)bidx * TOK + h * Wc + w) * Cc + gcol;
          out_f[idx] = add_src[idx] + v;
        } else {
          size_t idx = (size_t)grow * Cc + gcol;
          out_f[idx] = v + add_src[idx];
        }
      }
    }
  }
}

// --------------------------- WMMA attention --------------------------------
// One block per (window, head): S = Q K^T (scaled, +rel-bias, +shift-mask),
// softmax(f32), O = P V. 49 rows padded to 64. HEAD_DIM=32 == WMMA K exactly.
__global__ __launch_bounds__(256) void attn_wmma(
    const _Float16* __restrict__ qkv,     // [MROWS][576]
    const float* __restrict__ bias_table, // [169][6]
    _Float16* __restrict__ outh) {        // [MROWS][192]
  const int blk  = blockIdx.x;
  const int win  = blk / NHc;
  const int head = blk - win * NHc;
  const int wi   = win & (NWIN_IMG - 1);  // window index within image

  __shared__ __align__(32) _Float16 Qs[64 * 32];   // [n][d]
  __shared__ __align__(32) _Float16 Ks[64 * 32];   // [m][d]
  __shared__ __align__(32) _Float16 VsT[32 * 64];  // [d][m]  fragment-major
  __shared__ float    Ss[64 * 64];
  __shared__ __align__(32) _Float16 Ps[64 * 64];

  const int tid  = threadIdx.x;
  const int wave = tid >> 5;
  const int lane = tid & 31;
  const int nl   = lane & 15;
  const int hi   = lane >> 4;

  // ---- stage Q/K (vectorized, 8-half chunks) and V (transposed) ----
  {
    int r = tid >> 2, c = (tid & 3) * 8;          // 64 rows x 4 chunks
    v8h q = {}, k = {};
    if (r < Nwin) {
      size_t base = (size_t)(win * Nwin + r) * QKVN + head * HEADD + c;
      q = *(const v8h*)(qkv + base);
      k = *(const v8h*)(qkv + base + Cc);
    }
    *(v8h*)&Qs[r * 32 + c] = q;
    *(v8h*)&Ks[r * 32 + c] = k;
  }
  for (int idx = tid; idx < Nwin * 32; idx += 256) {    // V scatter-transpose
    int r = idx >> 5, d = idx & 31;
    VsT[d * 64 + r] = qkv[(size_t)(win * Nwin + r) * QKVN + head * HEADD + 2 * Cc + d];
  }
  for (int idx = tid; idx < 32 * 15; idx += 256) {      // zero pad cols 49..63
    int d = idx / 15, r = Nwin + idx % 15;
    VsT[d * 64 + r] = (_Float16)0.f;
  }
  __syncthreads();

  // ---- S = Q @ K^T : 4x4 grid of 16x16 tiles, 2 per wave ----
#pragma unroll
  for (int it = 0; it < 2; ++it) {
    int t  = wave * 2 + it;
    int ti = t >> 2, tj = t & 3;
    v16h af = frag_a(Qs + (ti * 16 + nl) * 32, hi);
    // B = K^T: column n is row n of K -> contiguous 16 halves
    v16h bf = *(const v16h*)(Ks + (tj * 16 + nl) * 32 + hi * 16);
    v8f sacc = {};
    sacc = WMMA16(af, bf, sacc);
#pragma unroll
    for (int r = 0; r < 8; ++r) {
      int qn = ti * 16 + r + hi * 8;  // query token
      int km = tj * 16 + nl;          // key token
      float sv = sacc[r] * SCALEF;
      if (qn < Nwin && km < Nwin) {
        int qh = qn / WSc, qw = qn % WSc;
        int kh = km / WSc, kw = km % WSc;
        sv += bias_table[((qh - kh + 6) * 13 + (qw - kw + 6)) * NHc + head];
        int hq = (wi >> 3) * WSc + qh, wq = (wi & 7) * WSc + qw;
        int hk = (wi >> 3) * WSc + kh, wk = (wi & 7) * WSc + kw;
        int lq = (hq < 49 ? 0 : (hq < 53 ? 1 : 2)) * 3 + (wq < 49 ? 0 : (wq < 53 ? 1 : 2));
        int lk = (hk < 49 ? 0 : (hk < 53 ? 1 : 2)) * 3 + (wk < 49 ? 0 : (wk < 53 ? 1 : 2));
        if (lq != lk) sv -= 100.0f;
      }
      Ss[qn * 64 + km] = sv;
    }
  }
  __syncthreads();

  for (int idx = tid; idx < 64 * 64; idx += 256) Ps[idx] = (_Float16)0.f;
  __syncthreads();

  if (tid < Nwin) {   // softmax over the 49 valid keys of row `tid`
    float m = -1e30f;
    for (int j = 0; j < Nwin; ++j) m = fmaxf(m, Ss[tid * 64 + j]);
    float s = 0.f;
    for (int j = 0; j < Nwin; ++j) s += expf(Ss[tid * 64 + j] - m);
    float inv = 1.f / s;
    for (int j = 0; j < Nwin; ++j)
      Ps[tid * 64 + j] = (_Float16)(expf(Ss[tid * 64 + j] - m) * inv);
  }
  __syncthreads();

  // ---- O = P @ V : 4x2 grid of 16x16 tiles, 1 per wave, K=64 in 2 steps ----
  int ti = wave & 3, tj = wave >> 2;
  v8f oacc = {};
#pragma unroll
  for (int ks = 0; ks < 64; ks += 32) {
    v16h af = frag_a(Ps + (ti * 16 + nl) * 64 + ks, hi);
    v16h bf = *(const v16h*)(VsT + (tj * 16 + nl) * 64 + ks + hi * 16);
    oacc = WMMA16(af, bf, oacc);
  }
#pragma unroll
  for (int r = 0; r < 8; ++r) {
    int qn = ti * 16 + r + hi * 8;
    int d  = tj * 16 + nl;
    if (qn < Nwin)
      outh[(size_t)(win * Nwin + qn) * Cc + head * HEADD + d] = (_Float16)oacc[r];
  }
}

// ---------------------------------------------------------------------------
extern "C" void kernel_launch(void* const* d_in, const int* in_sizes, int n_in,
                              void* d_out, int out_size, void* d_ws, size_t ws_size,
                              hipStream_t stream) {
  (void)in_sizes; (void)n_in; (void)out_size; (void)ws_size;

  const float* x      = (const float*)d_in[0];
  const float* n1g    = (const float*)d_in[1];
  const float* n1b    = (const float*)d_in[2];
  const float* qkv_w  = (const float*)d_in[3];
  const float* qkv_b  = (const float*)d_in[4];
  const float* relb   = (const float*)d_in[5];
  const float* proj_w = (const float*)d_in[6];
  const float* proj_b = (const float*)d_in[7];
  const float* n2g    = (const float*)d_in[8];
  const float* n2b    = (const float*)d_in[9];
  const float* fc1_w  = (const float*)d_in[10];
  const float* fc1_b  = (const float*)d_in[11];
  const float* fc2_w  = (const float*)d_in[12];
  const float* fc2_b  = (const float*)d_in[13];
  float* out = (float*)d_out;

  char* ws = (char*)d_ws;
  size_t off = 0;
  auto alloc = [&](size_t bytes) -> char* {
    char* p = ws + off;
    off = (off + bytes + 255) & ~(size_t)255;
    return p;
  };

  const size_t M = MROWS;
  _Float16* wqkvT = (_Float16*)alloc((size_t)Cc * QKVN * 2);     // [576][192]
  _Float16* wprojT= (_Float16*)alloc((size_t)Cc * Cc * 2);       // [192][192]
  _Float16* wfc1T = (_Float16*)alloc((size_t)Cc * HIDDEN * 2);   // [768][192]
  _Float16* wfc2T = (_Float16*)alloc((size_t)HIDDEN * Cc * 2);   // [192][768]
  _Float16* hwin  = (_Float16*)alloc(M * Cc * 2);      // LN1 windows; reused for attn out
  _Float16* qkvbuf= (_Float16*)alloc(M * QKVN * 2);    // qkv f16; reused for LN2 out
  float*    x1    = (float*)   alloc(M * Cc * 4);      // residual after attention
  _Float16* h3    = (_Float16*)alloc(M * HIDDEN * 2);  // fc1/gelu output

  _Float16* attnh = hwin;     // hwin is dead after the qkv GEMM
  _Float16* h2    = qkvbuf;   // qkv is dead after attention

  // 1) weights -> f16, transposed to [N][K] (fragment-major B for WMMA)
  cvt_transpose_f16<<<(Cc * QKVN + 255) / 256, 256, 0, stream>>>(qkv_w, wqkvT, Cc, QKVN);
  cvt_transpose_f16<<<(Cc * Cc + 255) / 256, 256, 0, stream>>>(proj_w, wprojT, Cc, Cc);
  cvt_transpose_f16<<<(Cc * HIDDEN + 255) / 256, 256, 0, stream>>>(fc1_w, wfc1T, Cc, HIDDEN);
  cvt_transpose_f16<<<(HIDDEN * Cc + 255) / 256, 256, 0, stream>>>(fc2_w, wfc2T, HIDDEN, Cc);

  // 2) LN1 + shift + window partition -> f16
  ln_kernel<<<MROWS / 8, 256, 0, stream>>>(x, n1g, n1b, hwin, 1);

  // 3) qkv GEMM: [M,192] x [192,576] -> f16
  gemm_wmma<<<dim3(MROWS / 128, QKVN / 64), 256, 0, stream>>>(
      hwin, wqkvT, qkv_b, MROWS, QKVN, Cc, 0, 0, qkvbuf, nullptr, nullptr);

  // 4) attention: one block per (window, head)
  attn_wmma<<<Bc * NWIN_IMG * NHc, 256, 0, stream>>>(qkvbuf, relb, attnh);

  // 5) proj GEMM + window reverse + un-shift + residual -> x1 (f32)
  gemm_wmma<<<dim3(MROWS / 128, Cc / 64), 256, 0, stream>>>(
      attnh, wprojT, proj_b, MROWS, Cc, Cc, 1, 0, nullptr, x1, x);

  // 6) LN2 -> f16
  ln_kernel<<<MROWS / 8, 256, 0, stream>>>(x1, n2g, n2b, h2, 0);

  // 7) fc1 GEMM + exact GELU -> f16
  gemm_wmma<<<dim3(MROWS / 128, HIDDEN / 64), 256, 0, stream>>>(
      h2, wfc1T, fc1_b, MROWS, HIDDEN, Cc, 0, 1, h3, nullptr, nullptr);

  // 8) fc2 GEMM + residual -> d_out (f32)
  gemm_wmma<<<dim3(MROWS / 128, Cc / 64), 256, 0, stream>>>(
      h3, wfc2T, fc2_b, MROWS, Cc, HIDDEN, 2, 0, nullptr, out, x1);
}
